// Kmeans_6133213299488
// MI455X (gfx1250) — matrix operand order repaired and checked by reading, not compiled, not run on hardware
//
#include <hip/hip_runtime.h>
#include <hip/hip_bf16.h>

typedef __attribute__((ext_vector_type(16))) _Float16 v16h;
typedef __attribute__((ext_vector_type(8)))  _Float16 v8h;
typedef __attribute__((ext_vector_type(4)))  _Float16 v4h;
typedef __attribute__((ext_vector_type(8)))  float    v8f;
typedef __attribute__((ext_vector_type(4)))  float    v4f;
typedef __attribute__((ext_vector_type(4)))  int      v4i;

#define B_SZ        4
#define L_SZ        4096
#define D_SZ        64
#define H_SZ        4
#define C_SZ        512
#define ROWS_PER_WG 128          // 8 waves * 16 rows
#define STRIDE      72           // padded f16 row stride (144B -> conflict-free b128)
#define CTILE       256          // clusters staged per pass (2 passes)
#define NCHUNK      (CTILE / 16) // 16 cluster-chunks per pass

// sched_group_barrier masks
#define SG_VALU    0x0002
#define SG_WMMA    0x0008
#define SG_DS_READ 0x0100

__device__ __forceinline__ v16h cat16(v8h lo, v8h hi) {
    return __builtin_shufflevector(lo, hi, 0,1,2,3,4,5,6,7,8,9,10,11,12,13,14,15);
}

__global__ __launch_bounds__(256)
void kmeans_assign_wmma(const float* __restrict__ x,
                        const float* __restrict__ means,
                        int* __restrict__ out)
{
    __shared__ _Float16 mtile[CTILE * STRIDE];        // 36864 B
    __shared__ _Float16 xtile[ROWS_PER_WG * STRIDE];  // 18432 B

    const int tid   = threadIdx.x;
    const int lane  = tid & 31;
    const int wave  = tid >> 5;
    const int wgRow = blockIdx.x * ROWS_PER_WG;
    const int h     = blockIdx.y;
    const int b     = blockIdx.z;

    // ---- stage + L2-normalize 16 rows per wave: fp32 global -> f16 LDS ----
    {
        const int r    = lane & 15;     // row within wave tile
        const int half = lane >> 4;     // columns 0..31 vs 32..63
        const int rowL = wave * 16 + r; // row within workgroup
        const float* xp = x + ((size_t)b * L_SZ + wgRow + rowL) * D_SZ + half * 32;
        v4f v[8];
        float ss = 0.f;
        #pragma unroll
        for (int i = 0; i < 8; ++i) {
            v[i] = *(const v4f*)(xp + i * 4);
            ss += v[i].x * v[i].x + v[i].y * v[i].y + v[i].z * v[i].z + v[i].w * v[i].w;
        }
        ss += __shfl_xor(ss, 16, 32);                     // combine column halves
        const float scale = 1.0f / fmaxf(sqrtf(ss), 1e-12f);
        _Float16* dst = &xtile[rowL * STRIDE + half * 32];
        #pragma unroll
        for (int i = 0; i < 8; ++i) {
            v4h hv;
            hv.x = (_Float16)(v[i].x * scale);
            hv.y = (_Float16)(v[i].y * scale);
            hv.z = (_Float16)(v[i].z * scale);
            hv.w = (_Float16)(v[i].w * scale);
            *(v4h*)(dst + i * 4) = hv;
        }
    }
    __syncthreads();

    // ---- load A fragments once (16x32 f16 layout; K=0..31 and K=32..63) ----
    const int r16 = lane & 15;
    const int k0  = (lane < 16) ? 0 : 16;
    const _Float16* ap = &xtile[(wave * 16 + r16) * STRIDE];
    const v16h a0 = cat16(*(const v8h*)(ap + k0),      *(const v8h*)(ap + k0 + 8));
    const v16h a1 = cat16(*(const v8h*)(ap + k0 + 32), *(const v8h*)(ap + k0 + 40));

    // two independent argmax trackers (even / odd chunk parity)
    float bv0[8], bv1[8];
    int   bi0[8], bi1[8];
    #pragma unroll
    for (int j = 0; j < 8; ++j) {
        bv0[j] = -__builtin_inff(); bi0[j] = 0;
        bv1[j] = -__builtin_inff(); bi1[j] = 0;
    }

    const int cn = lane & 15;              // cluster column within chunk (C/D: N = lane%16)
    const int kb = (lane < 16) ? 0 : 16;   // B fragment K base
    // per-lane base into the means tile; chunk offsets become immediates below
    const _Float16* const bbase = &mtile[cn * STRIDE + kb];

    #pragma unroll
    for (int pass = 0; pass < 2; ++pass) {
        __syncthreads();  // previous pass done reading mtile
        // ---- stage 256 cluster means: fp32 global -> f16 LDS ----
        const float* mh = means + ((size_t)h * C_SZ + pass * CTILE) * D_SZ;
        #pragma unroll 4
        for (int i = tid; i < CTILE * 16; i += 256) {
            const int cl = i >> 4, q = i & 15;
            v4f mv = *(const v4f*)(mh + cl * D_SZ + q * 4);
            v4h hv;
            hv.x = (_Float16)mv.x; hv.y = (_Float16)mv.y;
            hv.z = (_Float16)mv.z; hv.w = (_Float16)mv.w;
            *(v4h*)&mtile[cl * STRIDE + q * 4] = hv;
        }
        __syncthreads();

        // ---- fully unrolled, software-pipelined chunk loop ----
        // per region: prefetch next pair's 8 DS reads, 4 WMMAs on current pair,
        // then the pair's argmax VALU; enforced with sched_group_barrier.
        v16h b00 = cat16(*(const v8h*)(bbase + 0 * 16 * STRIDE),
                         *(const v8h*)(bbase + 0 * 16 * STRIDE + 8));
        v16h b01 = cat16(*(const v8h*)(bbase + 0 * 16 * STRIDE + 32),
                         *(const v8h*)(bbase + 0 * 16 * STRIDE + 40));
        v16h b10 = cat16(*(const v8h*)(bbase + 1 * 16 * STRIDE),
                         *(const v8h*)(bbase + 1 * 16 * STRIDE + 8));
        v16h b11 = cat16(*(const v8h*)(bbase + 1 * 16 * STRIDE + 32),
                         *(const v8h*)(bbase + 1 * 16 * STRIDE + 40));

        #pragma unroll
        for (int nc = 0; nc < NCHUNK; nc += 2) {
            v16h n00 = b00, n01 = b01, n10 = b10, n11 = b11;
            if (nc + 2 < NCHUNK) {   // prefetch next pair (immediate DS offsets)
                const _Float16* p2 = bbase + (nc + 2) * 16 * STRIDE;
                const _Float16* p3 = bbase + (nc + 3) * 16 * STRIDE;
                n00 = cat16(*(const v8h*)(p2),      *(const v8h*)(p2 + 8));
                n01 = cat16(*(const v8h*)(p2 + 32), *(const v8h*)(p2 + 40));
                n10 = cat16(*(const v8h*)(p3),      *(const v8h*)(p3 + 8));
                n11 = cat16(*(const v8h*)(p3 + 32), *(const v8h*)(p3 + 40));
            }

            v8f acc0 = {};
            acc0 = __builtin_amdgcn_wmma_f32_16x16x32_f16(false, a0, false, b00,
                                                          (short)0, acc0, false, false);
            acc0 = __builtin_amdgcn_wmma_f32_16x16x32_f16(false, a1, false, b01,
                                                          (short)0, acc0, false, false);
            v8f acc1 = {};
            acc1 = __builtin_amdgcn_wmma_f32_16x16x32_f16(false, a0, false, b10,
                                                          (short)0, acc1, false, false);
            acc1 = __builtin_amdgcn_wmma_f32_16x16x32_f16(false, a1, false, b11,
                                                          (short)0, acc1, false, false);

            const int c0 = pass * CTILE + nc * 16 + cn;
            const int c1 = c0 + 16;
            #pragma unroll
            for (int j = 0; j < 8; ++j) {
                const float v0 = acc0[j];
                if (v0 > bv0[j]) { bv0[j] = v0; bi0[j] = c0; }
            }
            #pragma unroll
            for (int j = 0; j < 8; ++j) {
                const float v1 = acc1[j];
                if (v1 > bv1[j]) { bv1[j] = v1; bi1[j] = c1; }
            }

            b00 = n00; b01 = n01; b10 = n10; b11 = n11;

            // prescribe the pipeline for this region:
            //   8 DS reads (next pair) -> 4 WMMAs (this pair) -> argmax VALU
            __builtin_amdgcn_sched_group_barrier(SG_DS_READ, 8, 0);
            __builtin_amdgcn_sched_group_barrier(SG_WMMA,    4, 0);
            __builtin_amdgcn_sched_group_barrier(SG_VALU,   48, 0);
        }
    }

    // ---- merge parity trackers (ties -> smaller cluster index) ----
    float bestv[8];
    int   besti[8];
    #pragma unroll
    for (int j = 0; j < 8; ++j) {
        if (bv1[j] > bv0[j] || (bv1[j] == bv0[j] && bi1[j] < bi0[j])) {
            bestv[j] = bv1[j]; besti[j] = bi1[j];
        } else {
            bestv[j] = bv0[j]; besti[j] = bi0[j];
        }
    }

    // ---- argmax reduction across the 16 lanes holding the same row ----
    #pragma unroll
    for (int j = 0; j < 8; ++j) {
        float v  = bestv[j];
        int   ix = besti[j];
        #pragma unroll
        for (int m = 1; m < 16; m <<= 1) {
            const float ov = __shfl_xor(v, m, 32);
            const int   oi = __shfl_xor(ix, m, 32);
            if (ov > v || (ov == v && oi < ix)) { v = ov; ix = oi; }
        }
        besti[j] = ix;
    }

    // lane 0 writes rows M=0..7 of the wave tile, lane 16 writes M=8..15
    if ((lane & 15) == 0) {
        const int    mbase = (lane >> 4) * 8;
        const size_t obase = ((size_t)b * H_SZ + h) * L_SZ + wgRow + wave * 16 + mbase;
        const int    off   = h * C_SZ;
        v4i o0, o1;
        o0.x = besti[0] + off; o0.y = besti[1] + off;
        o0.z = besti[2] + off; o0.w = besti[3] + off;
        o1.x = besti[4] + off; o1.y = besti[5] + off;
        o1.z = besti[6] + off; o1.w = besti[7] + off;
        *(v4i*)(out + obase)     = o0;   // 32B-aligned: obase is a multiple of 8
        *(v4i*)(out + obase + 4) = o1;
    }
}

extern "C" void kernel_launch(void* const* d_in, const int* in_sizes, int n_in,
                              void* d_out, int out_size, void* d_ws, size_t ws_size,
                              hipStream_t stream) {
    (void)in_sizes; (void)n_in; (void)d_ws; (void)ws_size; (void)out_size;
    const float* x     = (const float*)d_in[0];  // [4, 4096, 64] f32
    const float* means = (const float*)d_in[1];  // [4, 512, 64]  f32
    int* out = (int*)d_out;                      // [4, 4*4096] int32 codes

    dim3 grid(L_SZ / ROWS_PER_WG, H_SZ, B_SZ);   // (32, 4, 4)
    dim3 block(256);
    kmeans_assign_wmma<<<grid, block, 0, stream>>>(x, means, out);
}
